// ParallelFeatureNet_59339268161866
// MI455X (gfx1250) — matrix operand order repaired and checked by reading, not compiled, not run on hardware
//
#include <hip/hip_runtime.h>
#include <hip/hip_bf16.h>
#include <math.h>

// ---------- types ----------
typedef __attribute__((ext_vector_type(16))) __bf16          v16bf;
typedef __attribute__((ext_vector_type(8)))  float           v8f;
typedef __attribute__((ext_vector_type(8)))  unsigned short  v8s;

union ABFrag { v16bf bf; v8s s8[2]; };

// pack two f32 -> bf16x2: round-half-up (≈RNE) + one v_perm_b32 byte merge
__device__ __forceinline__ unsigned int pack_bf16x2(float v0, float v1) {
    const unsigned int u0 = __float_as_uint(v0) + 0x8000u;
    const unsigned int u1 = __float_as_uint(v1) + 0x8000u;
    // dst = { u1[31:24], u1[23:16], u0[31:24], u0[23:16] }  (src0=u1 -> bytes 4..7)
    return __builtin_amdgcn_perm(u1, u0, 0x07060302u);
}

#define HOUT 256
#define BM   128   // rows per block (8 waves x 16)
#define BN   64    // out-cols per block
#define BK   32    // k-step (one bf16 WMMA depth)
#define LDSS 40    // padded LDS row stride in halves: 80B -> 16B aligned, conflict-free
#define AIT  8     // A staging iterations per thread (BM*BK/2/256)
#define BIT  4     // B staging iterations per thread (BN*BK/2/256)

// Y[N,256] = X[N,K] @ W[256,K]^T  (+ optional SAGE epilogue).  K must be even.
// MODE 0: Y = gemm
// MODE 1: Y = gemm + bias + agg/max(deg,1)          (SAGE combine)
// MODE 2: like 1, but atomicAdd into pooled[batch[n]*512 + pooled_off + h] (fused mean-pool)
template <int MODE>
__global__ void __launch_bounds__(256)
gemm_wmma_bf16(const float* __restrict__ X, int N, int K,
               const float* __restrict__ W,      // [HOUT, K] row-major
               const float* __restrict__ bias,   // [HOUT]
               const float* __restrict__ agg,    // [N, HOUT]
               const float* __restrict__ deg,    // [N]
               const int*   __restrict__ batch,  // [N]
               float* __restrict__ Y,            // [N, HOUT]
               float* __restrict__ pooled,       // [G, 512]
               int pooled_off)
{
    __shared__ alignas(16) unsigned short As[BM * LDSS];
    __shared__ alignas(16) unsigned short Bs[BN * LDSS];

    const int tid  = threadIdx.x;
    const int wave = tid >> 5;       // 0..7
    const int lane = tid & 31;
    const int lh   = lane & 15;
    const int hi   = lane >> 4;

    const int rows0 = blockIdx.x * BM;
    const int cols0 = blockIdx.y * BN;
    const int Nm1   = N - 1;

    // ---- staging geometry (constant over K loop) ----
    const int sr_base = tid >> 4;        // row advances by 16 per staging iteration
    const int sc      = (tid & 15) * 2;  // bf16-pair offset within the 32-wide k tile

    // Row-clamped pointers: out-of-range A rows read row N-1 (harmless: row r of A
    // only affects row r of D, and those rows are never stored by the epilogue).
    const float* xrow[AIT];  int aoff[AIT];
    #pragma unroll
    for (int it = 0; it < AIT; ++it) {
        const int r = sr_base + it * 16;
        xrow[it] = X + (size_t)min(rows0 + r, Nm1) * K;
        aoff[it] = r * LDSS + sc;
    }
    const float* wrow[BIT];  int boff[BIT];
    #pragma unroll
    for (int it = 0; it < BIT; ++it) {
        const int h = sr_base + it * 16;
        wrow[it] = W + (size_t)(cols0 + h) * K;
        boff[it] = h * LDSS + sc;
    }

    // ---- software pipeline: prefetch global tile k+1 while computing tile k ----
    float2 pa[AIT], pb[BIT];

    auto prefetch = [&](int kk) {
        const int gk = kk + sc;
        const int cp = min(gk, K - 2);   // even (K even, sc even) -> aligned b64 loads
        #pragma unroll
        for (int it = 0; it < AIT; ++it) pa[it] = *(const float2*)&xrow[it][cp];
        #pragma unroll
        for (int it = 0; it < BIT; ++it) pb[it] = *(const float2*)&wrow[it][cp];
    };
    auto store_fast = [&]() {            // full K tile: no masking at all
        #pragma unroll
        for (int it = 0; it < AIT; ++it)
            *(unsigned int*)&As[aoff[it]] = pack_bf16x2(pa[it].x, pa[it].y);
        #pragma unroll
        for (int it = 0; it < BIT; ++it)
            *(unsigned int*)&Bs[boff[it]] = pack_bf16x2(pb[it].x, pb[it].y);
    };
    auto store_tail = [&](int kk) {      // partial last tile: zero the K overhang
        const bool kok = (kk + sc) < K;  // K even -> covers both pair elements
        #pragma unroll
        for (int it = 0; it < AIT; ++it) {
            const unsigned int pk = pack_bf16x2(pa[it].x, pa[it].y);
            *(unsigned int*)&As[aoff[it]] = kok ? pk : 0u;
        }
        #pragma unroll
        for (int it = 0; it < BIT; ++it) {
            const unsigned int pk = pack_bf16x2(pb[it].x, pb[it].y);
            *(unsigned int*)&Bs[boff[it]] = kok ? pk : 0u;
        }
    };

    v8f acc[4];
    #pragma unroll
    for (int t = 0; t < 4; ++t) acc[t] = {};

    const int abase  = (wave * 16 + lh) * LDSS + hi * 8;  // A frag: two 16B LDS loads
    const int bbase0 = lh * LDSS + hi * 8;                // B frag base for t=0

    prefetch(0);
    for (int kk = 0; kk < K; kk += BK) {
        __syncthreads();        // all waves done reading LDS from previous step
        if (kk + BK <= K) store_fast(); else store_tail(kk);
        __syncthreads();
        if (kk + BK < K) prefetch(kk + BK);   // next tile's loads fly during compute

        // load A fragment + all 4 B fragments, then burst 4 WMMAs
        ABFrag a, b[4];
        a.s8[0] = *(const v8s*)&As[abase];
        a.s8[1] = *(const v8s*)&As[abase + 16];
        #pragma unroll
        for (int t = 0; t < 4; ++t) {
            const int bb = bbase0 + t * 16 * LDSS;
            b[t].s8[0] = *(const v8s*)&Bs[bb];
            b[t].s8[1] = *(const v8s*)&Bs[bb + 16];
        }
        #pragma unroll
        for (int t = 0; t < 4; ++t)
            acc[t] = __builtin_amdgcn_wmma_f32_16x16x32_bf16(
                false, a.bf, false, b[t].bf, (short)0, acc[t], false, false);
    }

    // epilogue: D layout -> VGPR j: row = j + 8*hi, col = lh (per 16x16 tile)
#define EPILOGUE(GUARDED)                                                          \
    _Pragma("unroll")                                                              \
    for (int t = 0; t < 4; ++t) {                                                  \
        const int h = cols0 + t * 16 + lh;                                         \
        _Pragma("unroll")                                                          \
        for (int j = 0; j < 8; ++j) {                                              \
            const int n = rows0 + wave * 16 + hi * 8 + j;                          \
            if (!GUARDED || n < N) {                                               \
                float v = acc[t][j];                                               \
                if (MODE >= 1)                                                     \
                    v += bias[h] + agg[(size_t)n * HOUT + h] / fmaxf(deg[n], 1.0f);\
                if (MODE == 2)                                                     \
                    atomicAdd(&pooled[(size_t)batch[n] * 512 + pooled_off + h], v);\
                else                                                               \
                    Y[(size_t)n * HOUT + h] = v;                                   \
            }                                                                      \
        }                                                                          \
    }
    if (rows0 + BM <= N) { EPILOGUE(false) } else { EPILOGUE(true) }
#undef EPILOGUE
}

#define EDGES_PER_BLOCK 8
// agg[dst[e], :] += xl[src[e], :]
__global__ void __launch_bounds__(256)
scatter_edges(const float* __restrict__ xl, const int* __restrict__ src,
              const int* __restrict__ dst, float* __restrict__ agg, int E)
{
    const int t  = threadIdx.x;
    const int e0 = blockIdx.x * EDGES_PER_BLOCK;
    #pragma unroll
    for (int i = 0; i < EDGES_PER_BLOCK; ++i) {
        const int e = e0 + i;
        if (e < E) {
            const int s = src[e];
            const int d = dst[e];
            atomicAdd(&agg[(size_t)d * HOUT + t], xl[(size_t)s * HOUT + t]);
        }
    }
}

__global__ void __launch_bounds__(256)
degree_kernel(const int* __restrict__ dst, float* __restrict__ deg, int E)
{
    int i = blockIdx.x * 256 + threadIdx.x;
    if (i < E) atomicAdd(&deg[dst[i]], 1.0f);
}

__global__ void __launch_bounds__(256)
count_kernel(const int* __restrict__ batch, float* __restrict__ cnt, int N)
{
    int i = blockIdx.x * 256 + threadIdx.x;
    if (i < N) atomicAdd(&cnt[batch[i]], 1.0f);
}

// pooled mean -> lin1(512->256) relu -> lin2(256->2) -> log_softmax ; one block per graph
__global__ void __launch_bounds__(256)
mlp_head(const float* __restrict__ pooled, const float* __restrict__ cnt,
         const float* __restrict__ W1, const float* __restrict__ b1,
         const float* __restrict__ W2, const float* __restrict__ b2,
         float* __restrict__ out)
{
    __shared__ float prow[512];
    __shared__ float hbuf[256];
    __shared__ float obuf[2];
    const int g = blockIdx.x;
    const int t = threadIdx.x;

    const float c = fmaxf(cnt[g], 1.0f);
    for (int i = t; i < 512; i += 256) prow[i] = pooled[(size_t)g * 512 + i] / c;
    __syncthreads();

    float s = b1[t];
    for (int k = 0; k < 512; ++k) s += prow[k] * W1[(size_t)t * 512 + k];
    hbuf[t] = fmaxf(s, 0.0f);
    __syncthreads();

    if (t < 2) {
        float o = b2[t];
        for (int k = 0; k < 256; ++k) o += hbuf[k] * W2[(size_t)t * 256 + k];
        obuf[t] = o;
    }
    __syncthreads();

    if (t == 0) {
        float m = fmaxf(obuf[0], obuf[1]);
        float l = m + logf(expf(obuf[0] - m) + expf(obuf[1] - m));
        out[(size_t)g * 2 + 0] = obuf[0] - l;
        out[(size_t)g * 2 + 1] = obuf[1] - l;
    }
}

extern "C" void kernel_launch(void* const* d_in, const int* in_sizes, int n_in,
                              void* d_out, int out_size, void* d_ws, size_t ws_size,
                              hipStream_t stream) {
    const float* content_x = (const float*)d_in[0];
    const float* bert_x    = (const float*)d_in[1];
    const int*   eic       = (const int*)d_in[2];   // [2,E]: src row0, dst row1
    const int*   eib       = (const int*)d_in[3];
    const int*   batch     = (const int*)d_in[4];
    const float* Wl1_c = (const float*)d_in[5];  const float* bl1_c = (const float*)d_in[6];  const float* Wr1_c = (const float*)d_in[7];
    const float* Wl2_c = (const float*)d_in[8];  const float* bl2_c = (const float*)d_in[9];  const float* Wr2_c = (const float*)d_in[10];
    const float* Wl1_b = (const float*)d_in[11]; const float* bl1_b = (const float*)d_in[12]; const float* Wr1_b = (const float*)d_in[13];
    const float* Wl2_b = (const float*)d_in[14]; const float* bl2_b = (const float*)d_in[15]; const float* Wr2_b = (const float*)d_in[16];
    const float* W_lin1 = (const float*)d_in[17]; const float* b_lin1 = (const float*)d_in[18];
    const float* W_lin2 = (const float*)d_in[19]; const float* b_lin2 = (const float*)d_in[20];
    float* out = (float*)d_out;

    const int N  = in_sizes[4];
    const int E  = in_sizes[2] / 2;
    const int Kc = in_sizes[0] / N;   // 310
    const int Kb = in_sizes[1] / N;   // 768
    const int G  = out_size / 2;      // 128

    // workspace layout (floats)
    float* A      = (float*)d_ws;                  // [N,256] xl scratch
    float* Agg    = A    + (size_t)N * HOUT;       // [N,256] neighbor sums
    float* C      = Agg  + (size_t)N * HOUT;       // [N,256] layer-1 output
    float* degc   = C    + (size_t)N * HOUT;       // [N]
    float* degb   = degc + N;                      // [N]
    float* pooled = degb + N;                      // [G,512]
    float* cntg   = pooled + (size_t)G * 512;      // [G]

    const int* src_c = eic;     const int* dst_c = eic + E;
    const int* src_b = eib;     const int* dst_b = eib + E;

    dim3 ggrid((N + BM - 1) / BM, HOUT / BN);
    const int eBlocks = (E + 255) / 256;
    const int sBlocks = (E + EDGES_PER_BLOCK - 1) / EDGES_PER_BLOCK;
    const int nBlocks = (N + 255) / 256;
    const size_t aggBytes = (size_t)N * HOUT * sizeof(float);

    // one-time zeroing (capturable memset nodes)
    hipMemsetAsync(degc, 0, 2 * (size_t)N * sizeof(float), stream);
    hipMemsetAsync(pooled, 0, ((size_t)G * 512 + G) * sizeof(float), stream);

    degree_kernel<<<eBlocks, 256, 0, stream>>>(dst_c, degc, E);
    degree_kernel<<<eBlocks, 256, 0, stream>>>(dst_b, degb, E);
    count_kernel<<<nBlocks, 256, 0, stream>>>(batch, cntg, N);

    // ---- content branch ----
    gemm_wmma_bf16<0><<<ggrid, 256, 0, stream>>>(content_x, N, Kc, Wl1_c,
        nullptr, nullptr, nullptr, nullptr, A, nullptr, 0);
    hipMemsetAsync(Agg, 0, aggBytes, stream);
    scatter_edges<<<sBlocks, 256, 0, stream>>>(A, src_c, dst_c, Agg, E);
    gemm_wmma_bf16<1><<<ggrid, 256, 0, stream>>>(content_x, N, Kc, Wr1_c,
        bl1_c, Agg, degc, nullptr, C, nullptr, 0);

    gemm_wmma_bf16<0><<<ggrid, 256, 0, stream>>>(C, N, HOUT, Wl2_c,
        nullptr, nullptr, nullptr, nullptr, A, nullptr, 0);
    hipMemsetAsync(Agg, 0, aggBytes, stream);
    scatter_edges<<<sBlocks, 256, 0, stream>>>(A, src_c, dst_c, Agg, E);
    gemm_wmma_bf16<2><<<ggrid, 256, 0, stream>>>(C, N, HOUT, Wr2_c,
        bl2_c, Agg, degc, batch, nullptr, pooled, 0);      // fused mean-pool accumulate

    // ---- bert branch ----
    gemm_wmma_bf16<0><<<ggrid, 256, 0, stream>>>(bert_x, N, Kb, Wl1_b,
        nullptr, nullptr, nullptr, nullptr, A, nullptr, 0);
    hipMemsetAsync(Agg, 0, aggBytes, stream);
    scatter_edges<<<sBlocks, 256, 0, stream>>>(A, src_b, dst_b, Agg, E);
    gemm_wmma_bf16<1><<<ggrid, 256, 0, stream>>>(bert_x, N, Kb, Wr1_b,
        bl1_b, Agg, degb, nullptr, C, nullptr, 0);

    gemm_wmma_bf16<0><<<ggrid, 256, 0, stream>>>(C, N, HOUT, Wl2_b,
        nullptr, nullptr, nullptr, nullptr, A, nullptr, 0);
    hipMemsetAsync(Agg, 0, aggBytes, stream);
    scatter_edges<<<sBlocks, 256, 0, stream>>>(A, src_b, dst_b, Agg, E);
    gemm_wmma_bf16<2><<<ggrid, 256, 0, stream>>>(C, N, HOUT, Wr2_b,
        bl2_b, Agg, degb, batch, nullptr, pooled, 256);    // second half of concat

    // ---- head ----
    mlp_head<<<G, 256, 0, stream>>>(pooled, cntg, W_lin1, b_lin1, W_lin2, b_lin2, out);
}
